// TensorizedPC_86431921864929
// MI455X (gfx1250) — matrix operand order repaired and checked by reading, not compiled, not run on hardware
//
#include <hip/hip_runtime.h>
#include <hip/hip_bf16.h>
#include <math.h>

// ---------------------------------------------------------------------------
// TensorizedPC forward on gfx1250 (MI455X).
//   leaf:   log0[v,i,b] = -0.5*(x[b,v]-mu[v,i])^2 - 0.5*log(2pi)
//   layer:  out[f,o,b]  = LSE_i(wl[f,o,i]+in[2f,i,b]) + LSE_i(wr[f,o,i]+in[2f+1,i,b])
//           computed as log( exp(w-mw) @ exp(x-mx) ) + mw + mx  (stable)
//   The inner einsum runs on v_wmma_f32_16x16x32_f16 (f16 in, f32 accum):
//   after max-shift all operands are in (0,1], so f16 mantissa is adequate
//   and accumulation is full f32.
// ---------------------------------------------------------------------------

typedef __attribute__((ext_vector_type(16))) _Float16 v16h;
typedef __attribute__((ext_vector_type(8)))  _Float16 v8h;
typedef __attribute__((ext_vector_type(8)))  float    v8f;

#define HALF_LOG2PI 0.9189385332046727f
#define BATCH 256

// ---------------------------------------------------------------------------
// Leaf layer: one workgroup per leaf variable v. out: (1024, 64, 256) f32.
// ---------------------------------------------------------------------------
__global__ __launch_bounds__(256)
void pc_leaf_kernel(const float* __restrict__ x,    // (256, 1024)
                    const float* __restrict__ mu,   // (1024, 64)
                    float* __restrict__ out)        // (1024, 64, 256)
{
    const int v = blockIdx.x;
    const int b = threadIdx.x;
    __shared__ float xs[256];
    __shared__ float mus[64];
    xs[b] = x[(size_t)b * 1024 + v];
    if (b < 64) mus[b] = mu[(size_t)v * 64 + b];
    __syncthreads();
    const float xv = xs[b];
    const size_t base = (size_t)v * 64 * BATCH;
    #pragma unroll 4
    for (int i = 0; i < 64; ++i) {
        const float d = xv - mus[i];
        out[base + (size_t)i * BATCH + b] = -0.5f * d * d - HALF_LOG2PI;
    }
}

// ---------------------------------------------------------------------------
// Generic sum layer (O = 128). One workgroup per (f, 64-wide batch tile).
// 256 threads = 8 wave32; wave w owns output rows [16w, 16w+16).
// ---------------------------------------------------------------------------
template <int I>   // reduction length: 64 (layer 0) or 128 (layers 1..8)
__global__ __launch_bounds__(256)
void pc_layer_kernel(const float* __restrict__ in,   // (2F, I, 256)
                     const float* __restrict__ wl,   // (F, 128, I)
                     const float* __restrict__ wr,   // (F, 128, I)
                     float* __restrict__ out)        // (F, 128, 256)
{
    constexpr int IP = I + 8;                 // +16B pad: alignment + banks
    __shared__ __align__(16) _Float16 sA[128][IP];  // exp(w - mw), O x I
    __shared__ __align__(16) _Float16 sE[64][IP];   // exp(x - mx)^T, N x I
    __shared__ float sMW[128];
    __shared__ float sMX[64];

    const int f    = blockIdx.x >> 2;
    const int b0   = (blockIdx.x & 3) * 64;
    const int tid  = threadIdx.x;
    const int wave = tid >> 5;                // wave32
    const int lane = tid & 31;
    const int lm   = lane & 15;
    const int lh   = lane >> 4;               // lane half (0/1)
    const int m0   = wave * 16;               // this wave's O-row base

    float partial[4][8];                      // log-result, 4 N-tiles x 8 rows

    #pragma unroll
    for (int side = 0; side < 2; ++side) {
        const float* __restrict__ Wf = (side ? wr : wl) + (size_t)f * 128 * I;
        const float* __restrict__ Xf = in + (size_t)(2 * f + side) * I * BATCH;

        __syncthreads();   // previous side's LDS fully consumed

        // ---- per-row weight max (threads 0..127), per-col input max (128..191)
        if (tid < 128) {
            float m = -INFINITY;
            const float* row = Wf + (size_t)tid * I;
            #pragma unroll 8
            for (int i = 0; i < I; ++i) m = fmaxf(m, row[i]);
            sMW[tid] = m;
        } else if (tid < 192) {
            const int n = tid - 128;
            float m = -INFINITY;
            #pragma unroll 8
            for (int i = 0; i < I; ++i) m = fmaxf(m, Xf[(size_t)i * BATCH + b0 + n]);
            sMX[n] = m;
        }
        __syncthreads();

        // ---- stage exp(w - mw) as f16 (coalesced global reads)
        for (int idx = tid; idx < 128 * I; idx += 256) {
            const int o = idx / I, i = idx % I;
            sA[o][i] = (_Float16)__expf(Wf[idx] - sMW[o]);
        }
        // ---- stage exp(x - mx)^T as f16, N-major so K is contiguous per lane
        for (int idx = tid; idx < I * 64; idx += 256) {
            const int i = idx / 64, n = idx % 64;
            sE[n][i] = (_Float16)__expf(Xf[(size_t)i * BATCH + b0 + n] - sMX[n]);
        }
        __syncthreads();

        // ---- WMMA: S[o,b] = A @ E  (K = I), f32 accumulation
        v8f acc[4];
        v8f z = {};
        #pragma unroll
        for (int t = 0; t < 4; ++t) acc[t] = z;

        #pragma unroll
        for (int kb = 0; kb < I / 32; ++kb) {
            // A fragment (16-bit 16x32 layout): lane<16 -> K=0..7,16..23 ;
            // lane>=16 -> K=8..15,24..31 within this 32-K block.
            const int arow = m0 + lm;
            const int ak   = kb * 32 + lh * 8;
            const v8h a_lo = *(const v8h*)&sA[arow][ak];
            const v8h a_hi = *(const v8h*)&sA[arow][ak + 16];
            v16h afrag;
            #pragma unroll
            for (int j = 0; j < 8; ++j) { afrag[j] = a_lo[j]; afrag[j + 8] = a_hi[j]; }

            #pragma unroll
            for (int t = 0; t < 4; ++t) {
                // B fragment: lane<16 -> K=0..7,16..23 ; lane>=16 -> K=8..15,24..31, N = lane&15.
                const int n  = t * 16 + lm;
                const int ek = kb * 32 + lh * 8;
                const v8h e_lo = *(const v8h*)&sE[n][ek];
                const v8h e_hi = *(const v8h*)&sE[n][ek + 16];
                v16h efrag;
                #pragma unroll
                for (int j = 0; j < 8; ++j) { efrag[j] = e_lo[j]; efrag[j + 8] = e_hi[j]; }
                acc[t] = __builtin_amdgcn_wmma_f32_16x16x32_f16(
                    false, afrag, false, efrag, (short)0, acc[t], false, false);
            }
        }

        // ---- log + un-shift; C/D layout: M = lh*8 + r, N = lm
        #pragma unroll
        for (int t = 0; t < 4; ++t) {
            const float mx = sMX[t * 16 + lm];
            #pragma unroll
            for (int r = 0; r < 8; ++r) {
                const float s = fmaxf(acc[t][r], 1e-37f);   // guard f16 underflow
                const float vlog = __logf(s) + sMW[m0 + lh * 8 + r] + mx;
                if (side == 0) partial[t][r] = vlog;
                else           partial[t][r] += vlog;
            }
        }
    }

    // ---- store out[f, o, b]
    const size_t obase = (size_t)f * 128 * BATCH;
    #pragma unroll
    for (int t = 0; t < 4; ++t) {
        const int n = b0 + t * 16 + lm;
        #pragma unroll
        for (int r = 0; r < 8; ++r) {
            const int o = m0 + lh * 8 + r;
            out[obase + (size_t)o * BATCH + n] = partial[t][r];
        }
    }
}

// ---------------------------------------------------------------------------
// Final layer: F=1, O=1, I=128. out[b] = LSE(wl + left[:,b]) + LSE(wr + right[:,b]).
// ---------------------------------------------------------------------------
__global__ __launch_bounds__(256)
void pc_final_kernel(const float* __restrict__ in,  // (2, 128, 256)
                     const float* __restrict__ wl,  // (128,)
                     const float* __restrict__ wr,  // (128,)
                     float* __restrict__ out)       // (256,)
{
    __shared__ float sWl[128], sWr[128];
    const int b = threadIdx.x;
    if (b < 128) { sWl[b] = wl[b]; sWr[b] = wr[b]; }
    __syncthreads();

    float mwl = -INFINITY, mwr = -INFINITY;
    #pragma unroll 8
    for (int i = 0; i < 128; ++i) { mwl = fmaxf(mwl, sWl[i]); mwr = fmaxf(mwr, sWr[i]); }

    float mxl = -INFINITY, mxr = -INFINITY;
    #pragma unroll 8
    for (int i = 0; i < 128; ++i) {
        mxl = fmaxf(mxl, in[(size_t)i * BATCH + b]);
        mxr = fmaxf(mxr, in[(size_t)(128 + i) * BATCH + b]);
    }
    float sl = 0.f, sr = 0.f;
    #pragma unroll 8
    for (int i = 0; i < 128; ++i) {
        sl += __expf(sWl[i] - mwl) * __expf(in[(size_t)i * BATCH + b] - mxl);
        sr += __expf(sWr[i] - mwr) * __expf(in[(size_t)(128 + i) * BATCH + b] - mxr);
    }
    out[b] = __logf(sl) + mwl + mxl + __logf(sr) + mwr + mxr;
}

// ---------------------------------------------------------------------------
// Host launcher.
// Input flattening (dict order): d_in[0]=x, d_in[1]=mu,
// d_in[2..11]=w_left[0..9], d_in[12..21]=w_right[0..9].
// Workspace: two 64 MiB f32 ping-pong buffers (leaf out and layer outs).
// ---------------------------------------------------------------------------
extern "C" void kernel_launch(void* const* d_in, const int* in_sizes, int n_in,
                              void* d_out, int out_size, void* d_ws, size_t ws_size,
                              hipStream_t stream) {
    const float* x  = (const float*)d_in[0];
    const float* mu = (const float*)d_in[1];

    float* buf0 = (float*)d_ws;
    float* buf1 = buf0 + (size_t)16777216;   // 1024*64*256 == 512*128*256 elements

    // Leaf layer -> buf0 : (1024, 64, 256)
    pc_leaf_kernel<<<1024, 256, 0, stream>>>(x, mu, buf0);

    float* cur = buf0;
    float* nxt = buf1;

    // Layer 0: F=512, I=64
    pc_layer_kernel<64><<<512 * 4, 256, 0, stream>>>(
        cur, (const float*)d_in[2], (const float*)d_in[12], nxt);
    { float* t = cur; cur = nxt; nxt = t; }

    // Layers 1..8: I=128, F = 1024 >> (l+1)
    for (int l = 1; l <= 8; ++l) {
        const int F = 1024 >> (l + 1);
        pc_layer_kernel<128><<<F * 4, 256, 0, stream>>>(
            cur, (const float*)d_in[2 + l], (const float*)d_in[12 + l], nxt);
        { float* t = cur; cur = nxt; nxt = t; }
    }

    // Layer 9: F=1, O=1 -> d_out (256 floats)
    pc_final_kernel<<<1, 256, 0, stream>>>(
        cur, (const float*)d_in[11], (const float*)d_in[21], (float*)d_out);
}
